// FeatureMixerLayer_8830452761074
// MI455X (gfx1250) — matrix operand order repaired and checked by reading, not compiled, not run on hardware
//
#include <hip/hip_runtime.h>

// FeatureMixerLayer for MI455X (gfx1250): wave32 + v_wmma_f32_16x16x32_f16
// + async global->LDS staging (ASYNCcnt) for the spike-GEMM B tiles.
typedef __attribute__((ext_vector_type(16))) _Float16 v16h;
typedef __attribute__((ext_vector_type(8)))  float    v8f;

#define BB 128   // batch
#define CC 1024  // channels
#define DD 256   // in_dim = timesteps T = hidden O

__device__ __forceinline__ v8f wmma_f16(v16h a, v16h b, v8f c) {
  // (neg_a, A, neg_b, B, c_mod, C, reuse_a, reuse_b)
  return __builtin_amdgcn_wmma_f32_16x16x32_f16(false, a, false, b, (short)0, c, false, false);
}

// A-matrix (16x32 f16, row-major source, lda in elements).
// ISA layout: lane<16 -> row m=lane, K = {k0..k0+7, k0+16..k0+23};
//             lane>=16 -> row m=lane-16, K = {k0+8..k0+15, k0+24..k0+31}
__device__ __forceinline__ v16h load_a_h(const _Float16* A, int lda, int m0, int k0, int lane) {
  const _Float16* p = A + (long)(m0 + (lane & 15)) * lda + (k0 + ((lane & 16) ? 8 : 0));
  v16h r;
#pragma unroll
  for (int i = 0; i < 8; ++i) { r[i] = p[i]; r[i + 8] = p[i + 16]; }
  return r;
}

// B-matrix (32x16 f16). Source stored [n][k] (column n contiguous in k), ldb in elems.
// ISA layout: lane<16 -> col n=lane, K = {k0..k0+7, k0+16..k0+23}; lane>=16 -> col n=lane-16, K = {k0+8..k0+15, k0+24..k0+31}
__device__ __forceinline__ v16h load_b_h(const _Float16* Bm, int ldb, int n0, int k0, int lane) {
  const _Float16* p = Bm + (long)(n0 + (lane & 15)) * ldb + (k0 + ((lane & 16) ? 8 : 0));
  v16h r;
#pragma unroll
  for (int i = 0; i < 8; ++i) { r[i] = p[i]; r[i + 8] = p[i + 16]; }
  return r;
}

// ---- CDNA5 async global->LDS copy (16B per lane per issue, ASYNCcnt) ----------
// ISA §15.18: same IOFFSET both sides, so a
// single (lds_off, gaddr) pair stepped by 16B is correct.
__device__ __forceinline__ void async_copy_b128(unsigned lds_off, const void* gaddr) {
  asm volatile("global_load_async_to_lds_b128 %0, %1, off"
               :: "v"(lds_off), "v"(gaddr)
               : "memory");
}
__device__ __forceinline__ void async_wait0() {
  asm volatile("s_wait_asynccnt 0x0" ::: "memory");
}

// ---------------- kernel 1: convert weights to f16 ----------------
__global__ __launch_bounds__(256) void cvt_kernel(
    const float* w1, const float* w2, const float* cw1, const float* cw2,
    _Float16* w1h, _Float16* w2h, _Float16* cw1h, _Float16* cw2h) {
  int i = blockIdx.x * 256 + threadIdx.x;
  if (i < DD * DD) { w1h[i] = (_Float16)w1[i]; w2h[i] = (_Float16)w2[i]; }
  if (i < DD * CC) { cw1h[i] = (_Float16)cw1[i]; cw2h[i] = (_Float16)cw2[i]; }
}

// ---------------- kernel 2: fused LN -> Linear -> ReLU -> Linear + residual ----
// One block = 16 rows of the [B*C, D] matrix. 256 threads = 8 waves.
__global__ __launch_bounds__(256) void mix_kernel(
    const float* __restrict__ x, const float* __restrict__ ln_g, const float* __restrict__ ln_b,
    const _Float16* __restrict__ w1h, const float* __restrict__ b1,
    const _Float16* __restrict__ w2h, const float* __restrict__ b2,
    float* __restrict__ out) {
  __shared__ float    xs[16][DD];
  __shared__ _Float16 xnh[16][DD];
  __shared__ _Float16 hs[16][DD];
  __shared__ float    red[2][16][16];
  __shared__ float    mu_s[16], rs_s[16];

  const int  tid = threadIdx.x;
  const long r0  = (long)blockIdx.x * 16;
  const int  row = tid >> 4, seg = tid & 15;

  // load 16x256 tile (each thread: 16 contiguous floats), partial LN sums
  const float* xrow = x + (r0 + row) * DD + seg * 16;
  float psum = 0.f, psq = 0.f;
#pragma unroll
  for (int j = 0; j < 16; ++j) {
    float v = xrow[j];
    xs[row][seg * 16 + j] = v;
    psum += v; psq += v * v;
  }
  red[0][row][seg] = psum; red[1][row][seg] = psq;
  __syncthreads();
  if (tid < 16) {
    float s = 0.f, q = 0.f;
#pragma unroll
    for (int j = 0; j < 16; ++j) { s += red[0][tid][j]; q += red[1][tid][j]; }
    float mu  = s * (1.f / DD);
    float var = q * (1.f / DD) - mu * mu;
    mu_s[tid] = mu;
    rs_s[tid] = __frsqrt_rn(var + 1e-5f);
  }
  __syncthreads();
  {
    float mu = mu_s[row], rs = rs_s[row];
#pragma unroll
    for (int j = 0; j < 16; ++j) {
      int col = seg * 16 + j;
      xnh[row][col] = (_Float16)((xs[row][col] - mu) * rs * ln_g[col] + ln_b[col]);
    }
  }
  __syncthreads();

  const int wave = tid >> 5, lane = tid & 31;

  // GEMM1: h = relu(xn @ w1^T + b1); each wave does 2 of the 16 N-tiles
#pragma unroll
  for (int tt = 0; tt < 2; ++tt) {
    int n0 = (wave * 2 + tt) * 16;
    v8f acc = {};
#pragma unroll
    for (int k0 = 0; k0 < DD; k0 += 32) {
      v16h a  = load_a_h(&xnh[0][0], DD, 0, k0, lane);
      v16h bf = load_b_h(w1h, DD, n0, k0, lane);   // B(d,e)=w1[e*D+d]: n-major contiguous-k
      acc = wmma_f16(a, bf, acc);
    }
    int n = n0 + (lane & 15);
    float bias = b1[n];
    int mbase = (lane & 16) ? 8 : 0;
#pragma unroll
    for (int i = 0; i < 8; ++i) {
      float v = acc[i] + bias;
      hs[mbase + i][n] = (_Float16)(v > 0.f ? v : 0.f);
    }
  }
  __syncthreads();

  // GEMM2: out = x + h @ w2^T + b2
#pragma unroll
  for (int tt = 0; tt < 2; ++tt) {
    int n0 = (wave * 2 + tt) * 16;
    v8f acc = {};
#pragma unroll
    for (int k0 = 0; k0 < DD; k0 += 32) {
      v16h a  = load_a_h(&hs[0][0], DD, 0, k0, lane);
      v16h bf = load_b_h(w2h, DD, n0, k0, lane);   // B(e,d)=w2[d*D+e]
      acc = wmma_f16(a, bf, acc);
    }
    int n = n0 + (lane & 15);
    float bias = b2[n];
    int mbase = (lane & 16) ? 8 : 0;
#pragma unroll
    for (int i = 0; i < 8; ++i) {
      int m = mbase + i;
      out[(r0 + m) * DD + n] = xs[m][n] + acc[i] + bias;
    }
  }
}

// ---------------- kernel 3: LIF over t on x[b,c,:]; spikes -> s1 [T][B][C] f16 ---
__global__ __launch_bounds__(256) void lif1_kernel(const float* __restrict__ x,
                                                   _Float16* __restrict__ s1) {
  long idx = (long)blockIdx.x * 256 + threadIdx.x;  // idx = b*C + c
  if (idx >= (long)BB * CC) return;
  const float* xp = x + idx * DD;                   // contiguous over t
  long b = idx / CC, c = idx % CC;
  _Float16* sp = s1 + b * CC + c;
  float v = 0.f;
  for (int t = 0; t < DD; ++t) {
    v = (v + xp[t]) * 0.5f;                         // v += (x - v)/tau, tau=2
    float s = (v >= 1.0f) ? 1.0f : 0.0f;
    v = (1.0f - s) * v;                             // hard reset
    sp[(long)t * BB * CC] = (_Float16)s;
  }
}

// ---------------- kernel 4: u[t,o,b] = cw1 @ s1[t] + cb1, stored as u[O][B][T] ---
// One block = one (t, b-tile). B tile (16 x 1024 f16 = 32 KB) staged in LDS via
// async-to-LDS; 8 waves each compute two 16x16 o-tiles (all 16 o-tiles covered).
__global__ __launch_bounds__(256) void sgemm1_kernel(const _Float16* __restrict__ cw1h,
                                                     const _Float16* __restrict__ s1,
                                                     const float* __restrict__ cb1,
                                                     float* __restrict__ u) {
  __shared__ _Float16 bsh[16][CC];                  // [b-col][c] : 32 KB
  const int tid = threadIdx.x;
  const int blk = blockIdx.x;                       // t*8 + bT : 2048 blocks
  const int t = blk >> 3, b0 = (blk & 7) * 16;

  // stage B tile: 16 rows x 2048B; 256 threads x 128B (8 async b128 each)
  {
    int r = tid >> 4, sg = (tid & 15) * 128;
    const char* g = (const char*)(s1 + ((long)t * BB + (b0 + r)) * CC) + sg;
    unsigned l = (unsigned)(size_t)(&bsh[r][0]) + sg;
#pragma unroll
    for (int i = 0; i < 8; ++i) async_copy_b128(l + 16 * i, g + 16 * i);
  }
  async_wait0();
  __syncthreads();

  const int wave = tid >> 5, lane = tid & 31;
  const int o0 = wave * 32;                         // this wave: o-tiles o0, o0+16
  v8f acc0 = {}, acc1 = {};
  for (int k0 = 0; k0 < CC; k0 += 32) {
    v16h bf = load_b_h(&bsh[0][0], CC, 0, k0, lane);
    v16h a0 = load_a_h(cw1h, CC, o0, k0, lane);
    v16h a1 = load_a_h(cw1h, CC, o0 + 16, k0, lane);
    acc0 = wmma_f16(a0, bf, acc0);
    acc1 = wmma_f16(a1, bf, acc1);
  }
  int b = b0 + (lane & 15);
  int mb = (lane & 16) ? 8 : 0;
#pragma unroll
  for (int i = 0; i < 8; ++i) {
    int oa = o0 + mb + i;
    int ob = o0 + 16 + mb + i;
    u[((long)oa * BB + b) * DD + t] = acc0[i] + cb1[oa];
    u[((long)ob * BB + b) * DD + t] = acc1[i] + cb1[ob];
  }
}

// ---------------- BN stats: one block per channel; slab is contiguous ----------
__global__ __launch_bounds__(256) void bnstats_kernel(const float* __restrict__ y,
                                                      float* __restrict__ mu,
                                                      float* __restrict__ rs, int count) {
  __shared__ float sS[256], sQ[256];
  const float* p = y + (long)blockIdx.x * count;
  float s = 0.f, q = 0.f;
  for (int i = threadIdx.x; i < count; i += 256) { float v = p[i]; s += v; q += v * v; }
  sS[threadIdx.x] = s; sQ[threadIdx.x] = q;
  __syncthreads();
  for (int off = 128; off > 0; off >>= 1) {
    if (threadIdx.x < off) { sS[threadIdx.x] += sS[threadIdx.x + off];
                             sQ[threadIdx.x] += sQ[threadIdx.x + off]; }
    __syncthreads();
  }
  if (threadIdx.x == 0) {
    float m = sS[0] / count;
    float var = sQ[0] / count - m * m;
    mu[blockIdx.x] = m;
    rs[blockIdx.x] = __frsqrt_rn(var + 1e-5f);
  }
}

// ---------------- kernel 6: BN1 + LIF2 over t; spikes -> s2 [T][B][O] f16 -------
__global__ __launch_bounds__(256) void lif2_kernel(const float* __restrict__ u,
                                                   const float* __restrict__ mu1,
                                                   const float* __restrict__ rs1,
                                                   const float* __restrict__ g1,
                                                   const float* __restrict__ be1,
                                                   _Float16* __restrict__ s2) {
  int idx = blockIdx.x * 256 + threadIdx.x;         // o*B + b
  if (idx >= DD * BB) return;
  int o = idx >> 7, b = idx & (BB - 1);
  const float* up = u + (long)idx * DD;             // contiguous over t
  float m = mu1[o], r = rs1[o], g = g1[o], be = be1[o];
  _Float16* sp = s2 + b * DD + o;
  float v = 0.f;
  for (int t = 0; t < DD; ++t) {
    float y = (up[t] - m) * r * g + be;
    v = (v + y) * 0.5f;
    float s = (v >= 1.0f) ? 1.0f : 0.0f;
    v = (1.0f - s) * v;
    sp[(long)t * BB * DD] = (_Float16)s;
  }
}

// ---------------- kernel 7: v2[t,c,b] = cw2 @ s2[t] + cb2, stored v2[C][B][T] ---
// One block = one (t, b-tile). B tile (16 x 256 f16 = 8 KB) staged in LDS via
// async-to-LDS; 8 waves each compute eight 16x16 c-tiles (all 64 covered),
// reusing each B fragment across 8 WMMA issues.
__global__ __launch_bounds__(256) void sgemm2_kernel(const _Float16* __restrict__ cw2h,
                                                     const _Float16* __restrict__ s2,
                                                     const float* __restrict__ cb2,
                                                     float* __restrict__ v2) {
  __shared__ _Float16 bsh[16][DD];                  // [b-col][o] : 8 KB
  const int tid = threadIdx.x;
  const int blk = blockIdx.x;                       // t*8 + bT : 2048 blocks
  const int t = blk >> 3, b0 = (blk & 7) * 16;

  // stage B tile: 16 rows x 512B; 256 threads x 32B (2 async b128 each)
  {
    int r = tid >> 4, sg = (tid & 15) * 32;
    const char* g = (const char*)(s2 + ((long)t * BB + (b0 + r)) * DD) + sg;
    unsigned l = (unsigned)(size_t)(&bsh[r][0]) + sg;
    async_copy_b128(l, g);
    async_copy_b128(l + 16, g + 16);
  }
  async_wait0();
  __syncthreads();

  const int wave = tid >> 5, lane = tid & 31;
  const int c0 = wave * 128;                        // this wave: c-tiles c0 + 16*j
  v8f acc[8] = {};
  for (int k0 = 0; k0 < DD; k0 += 32) {
    v16h bf = load_b_h(&bsh[0][0], DD, 0, k0, lane);
#pragma unroll
    for (int j = 0; j < 8; ++j) {
      v16h a = load_a_h(cw2h, DD, c0 + 16 * j, k0, lane);
      acc[j] = wmma_f16(a, bf, acc[j]);
    }
  }
  int b = b0 + (lane & 15);
  int mb = (lane & 16) ? 8 : 0;
#pragma unroll
  for (int j = 0; j < 8; ++j) {
#pragma unroll
    for (int i = 0; i < 8; ++i) {
      int c = c0 + 16 * j + mb + i;
      v2[((long)c * BB + b) * DD + t] = acc[j][i] + cb2[c];
    }
  }
}

// ---------------- kernel 9: out[b,c,t] += BN2(v2[c,b,t]) ------------------------
__global__ __launch_bounds__(256) void final_kernel(const float* __restrict__ v2,
                                                    const float* __restrict__ mu2,
                                                    const float* __restrict__ rs2,
                                                    const float* __restrict__ g2,
                                                    const float* __restrict__ be2,
                                                    float* __restrict__ out) {
  long idx = (long)blockIdx.x * 256 + threadIdx.x;  // b*C*T + c*T + t
  if (idx >= (long)BB * CC * DD) return;
  int t = (int)(idx & (DD - 1));
  long bc = idx >> 8;
  int c = (int)(bc % CC);
  long b = bc / CC;
  float y = (v2[((long)c * BB + b) * DD + t] - mu2[c]) * rs2[c] * g2[c] + be2[c];
  out[idx] += y;
}

extern "C" void kernel_launch(void* const* d_in, const int* in_sizes, int n_in,
                              void* d_out, int out_size, void* d_ws, size_t ws_size,
                              hipStream_t stream) {
  (void)in_sizes; (void)n_in; (void)out_size; (void)ws_size;
  const float* x    = (const float*)d_in[0];
  const float* ln_g = (const float*)d_in[1];
  const float* ln_b = (const float*)d_in[2];
  const float* w1   = (const float*)d_in[3];
  const float* b1   = (const float*)d_in[4];
  const float* w2   = (const float*)d_in[5];
  const float* b2   = (const float*)d_in[6];
  const float* cw1  = (const float*)d_in[7];
  const float* cb1  = (const float*)d_in[8];
  const float* bn1g = (const float*)d_in[9];
  const float* bn1b = (const float*)d_in[10];
  const float* cw2  = (const float*)d_in[11];
  const float* cb2  = (const float*)d_in[12];
  const float* bn2g = (const float*)d_in[13];
  const float* bn2b = (const float*)d_in[14];
  float* out = (float*)d_out;

  char* ws = (char*)d_ws;
  _Float16* w1h  = (_Float16*)(ws + 0);
  _Float16* w2h  = (_Float16*)(ws + 131072);
  _Float16* cw1h = (_Float16*)(ws + 262144);
  _Float16* cw2h = (_Float16*)(ws + 786432);
  float* mu1 = (float*)(ws + 1310720);
  float* rs1 = (float*)(ws + 1310720 + 1024);
  float* mu2 = (float*)(ws + 1310720 + 2048);
  float* rs2 = (float*)(ws + 1310720 + 2048 + 4096);
  _Float16* s1 = (_Float16*)(ws + (size_t)2  * 1024 * 1024);   //  64 MB [T][B][C]
  float*    u  = (float*)   (ws + (size_t)70 * 1024 * 1024);   //  32 MB [O][B][T]
  _Float16* s2 = (_Float16*)(ws + (size_t)103 * 1024 * 1024);  //  16 MB [T][B][O]
  float*    v2 = (float*)   (ws + (size_t)120 * 1024 * 1024);  // 128 MB [C][B][T]

  cvt_kernel<<<(DD * CC + 255) / 256, 256, 0, stream>>>(w1, w2, cw1, cw2, w1h, w2h, cw1h, cw2h);
  mix_kernel<<<(BB * CC) / 16, 256, 0, stream>>>(x, ln_g, ln_b, w1h, b1, w2h, b2, out);
  lif1_kernel<<<(BB * CC) / 256, 256, 0, stream>>>(x, s1);
  sgemm1_kernel<<<DD * (BB / 16), 256, 0, stream>>>(cw1h, s1, cb1, u);   // 2048 blocks
  bnstats_kernel<<<DD, 256, 0, stream>>>(u, mu1, rs1, BB * DD);          // per-o over (t,b)
  lif2_kernel<<<(DD * BB) / 256, 256, 0, stream>>>(u, mu1, rs1, bn1g, bn1b, s2);
  sgemm2_kernel<<<DD * (BB / 16), 256, 0, stream>>>(cw2h, s2, cb2, v2);  // 2048 blocks
  bnstats_kernel<<<CC, 256, 0, stream>>>(v2, mu2, rs2, BB * DD);         // per-c over (t,b)
  final_kernel<<<((long)BB * CC * DD) / 256, 256, 0, stream>>>(v2, mu2, rs2, bn2g, bn2b, out);
}
